// RoPEMultiheadSelfAttention_35777077576539
// MI455X (gfx1250) — matrix operand order repaired and checked by reading, not compiled, not run on hardware
//
#include <hip/hip_runtime.h>
#include <stdint.h>

#define BSZ 4
#define SEQ 2048
#define DM 1024
#define NH 16
#define HD 64
#define ROWS (BSZ * SEQ)   // 8192
#define NQKV (3 * DM)      // 3072

typedef _Float16 f16_t;
typedef f16_t v16h __attribute__((ext_vector_type(16)));
typedef float v8f __attribute__((ext_vector_type(8)));

// ---- WMMA fragment index helpers (wave32, 16x16x32 f16) -------------------
// A (16x32, lane = row M): VGPR j holds f16 pair at K = kpairA(j, half)
__device__ __forceinline__ int kpairA(int j, int half) {
  return ((j >> 2) << 4) + (half << 3) + ((j & 3) << 1);
}
// B (32x16, lane = col N): VGPR j holds f16 pair at K = kpairB(j, half)
__device__ __forceinline__ int kpairB(int j, int half) {
  return ((j >> 2) << 4) + (half << 3) + ((j & 3) << 1);
}

union FragU { uint32_t u[8]; v16h v; };

__device__ __forceinline__ v16h load_fragA(const uint16_t* base, int half) {
  FragU f;
#pragma unroll
  for (int j = 0; j < 8; ++j)
    f.u[j] = *reinterpret_cast<const uint32_t*>(base + kpairA(j, half));
  return f.v;
}
__device__ __forceinline__ v16h load_fragB(const uint16_t* base, int half) {
  FragU f;
#pragma unroll
  for (int j = 0; j < 8; ++j)
    f.u[j] = *reinterpret_cast<const uint32_t*>(base + kpairB(j, half));
  return f.v;
}

__device__ __forceinline__ v8f wmma16(v16h a, v16h b, v8f c) {
  return __builtin_amdgcn_wmma_f32_16x16x32_f16(false, a, false, b, (short)0, c,
                                                false, false);
}

__device__ __forceinline__ uint16_t f32_to_f16b(float f) {
  union { f16_t h; uint16_t u; } cv;
  cv.h = (f16_t)f;
  return cv.u;
}

// CDNA5 async global->LDS copy, 16B per lane. VDST VGPR carries the LDS byte
// address (low 32 bits of a generic LDS pointer == LDS offset per aperture
// rules); tracked on ASYNCcnt.
__device__ __forceinline__ void async_load_b128(void* lds_dst,
                                                const void* gsrc) {
  uint32_t laddr = (uint32_t)(uintptr_t)lds_dst;
  asm volatile("global_load_async_to_lds_b128 %0, %1, off"
               :
               : "v"(laddr), "v"(gsrc)
               : "memory");
}
__device__ __forceinline__ void wait_async0() {
  asm volatile("s_wait_asynccnt 0x0" ::: "memory");
}

__device__ __forceinline__ float half_reduce_max(float v) {
  v = fmaxf(v, __shfl_xor(v, 1, 32));
  v = fmaxf(v, __shfl_xor(v, 2, 32));
  v = fmaxf(v, __shfl_xor(v, 4, 32));
  v = fmaxf(v, __shfl_xor(v, 8, 32));
  return v;
}
__device__ __forceinline__ float half_reduce_sum(float v) {
  v += __shfl_xor(v, 1, 32);
  v += __shfl_xor(v, 2, 32);
  v += __shfl_xor(v, 4, 32);
  v += __shfl_xor(v, 8, 32);
  return v;
}

// ---- Kernel 0: fp32 -> f16 convert ----------------------------------------
__global__ __launch_bounds__(256) void cvt_f32_f16(const float* __restrict__ in,
                                                   uint16_t* __restrict__ out,
                                                   int n) {
  int i = blockIdx.x * 256 + threadIdx.x;
  if (i < n) out[i] = f32_to_f16b(in[i]);
}

// ---- Kernel 1: QKV GEMM + bias + RoPE + layout shuffle --------------------
// C[8192][3072] = xb @ wq^T ; 32x64 C tile per wave (2 A-frags x 4 B-frags).
// Writes Q,K (post-RoPE, Q pre-scaled by 1/8) as [b,h,s,d] f16 and V
// transposed [b,h,d,s] f16.
__global__ __launch_bounds__(128) void qkv_gemm_rope(
    const uint16_t* __restrict__ xb, const uint16_t* __restrict__ wq,
    const float* __restrict__ bias, uint16_t* __restrict__ qb,
    uint16_t* __restrict__ kb, uint16_t* __restrict__ vT) {
  const int lane = threadIdx.x & 31;
  const int wave = threadIdx.x >> 5;
  const int half = lane >> 4;
  const int r = lane & 15;
  const int tile = blockIdx.x * 4 + wave;
  const int NT = NQKV / 64;  // 48 column tiles of 64
  const int mt = tile / NT, nt = tile % NT;
  const int m0 = mt * 32, n0 = nt * 64;

  v8f acc[2][4];
#pragma unroll
  for (int p = 0; p < 2; ++p)
#pragma unroll
    for (int t = 0; t < 4; ++t)
#pragma unroll
      for (int e = 0; e < 8; ++e) acc[p][t][e] = 0.f;

  const uint16_t* arow0 = xb + (size_t)(m0 + r) * DM;
  const uint16_t* arow1 = xb + (size_t)(m0 + 16 + r) * DM;
  for (int k0 = 0; k0 < DM; k0 += 32) {
    __builtin_prefetch(arow0 + k0 + 32, 0, 0);
    __builtin_prefetch(arow1 + k0 + 32, 0, 0);
    v16h a0 = load_fragA(arow0 + k0, half);
    v16h a1 = load_fragA(arow1 + k0, half);
#pragma unroll
    for (int t = 0; t < 4; ++t) {
      v16h b = load_fragB(wq + (size_t)(n0 + t * 16 + r) * DM + k0, half);
      acc[0][t] = wmma16(a0, b, acc[0][t]);
      acc[1][t] = wmma16(a1, b, acc[1][t]);
    }
  }

  const int part = n0 / DM;        // 0=q 1=k 2=v (64-tile never straddles)
  const int head = (n0 % DM) / HD; // 64-tile == exactly one head
  const float LN1E4_D32 = 0.28782313662425572f;  // ln(10000)/32
#pragma unroll
  for (int p = 0; p < 2; ++p) {
#pragma unroll
    for (int t = 0; t < 4; ++t) {
      const int d = t * 16 + r;
      const float bv = bias[n0 + t * 16 + r];
#pragma unroll
      for (int i = 0; i < 8; ++i) acc[p][t][i] += bv;
      if (part < 2) {
        const float theta = __expf(-(float)(d >> 1) * LN1E4_D32);
#pragma unroll
        for (int i = 0; i < 8; ++i) {
          const int row = m0 + p * 16 + i + 8 * half;
          const int s = row & (SEQ - 1), b = row >> 11;
          float c, sn;
          __sincosf((float)s * theta, &sn, &c);
          const float v = acc[p][t][i];
          const float pr = __shfl_xor(v, 1, 32);  // partner dim d^1
          float o = (d & 1) ? (v * c + pr * sn) : (v * c - pr * sn);
          if (part == 0) o *= 0.125f;  // fold 1/sqrt(HD) into Q
          const size_t idx = ((size_t)((b * NH + head) * SEQ + s)) * HD + d;
          (part == 0 ? qb : kb)[idx] = f32_to_f16b(o);
        }
      } else {
#pragma unroll
        for (int i = 0; i < 8; ++i) {
          const int row = m0 + p * 16 + i + 8 * half;
          const int s = row & (SEQ - 1), b = row >> 11;
          vT[((size_t)(b * NH + head) * HD + d) * SEQ + s] =
              f32_to_f16b(acc[p][t][i]);
        }
      }
    }
  }
}

// ---- Kernel 2: flash attention --------------------------------------------
// 4 waves/block share one (b,h). K/V tiles are double-buffered in LDS and
// filled with CDNA5 async global->LDS loads so the next tile's DMA overlaps
// the current tile's WMMAs; one barrier per 32-key step.
__global__ __launch_bounds__(128) void flash_attn(
    const uint16_t* __restrict__ qb, const uint16_t* __restrict__ kb,
    const uint16_t* __restrict__ vT, uint16_t* __restrict__ ob) {
  __shared__ uint16_t Kt[2][32][64];     // 32 keys x 64 dims   (2 x 4 KB)
  __shared__ uint16_t Vt[2][64][32];     // 64 dims x 32 keys   (2 x 4 KB)
  __shared__ uint16_t Plds[4][16][32];   // per-wave P bounce   (4 KB)
  const int tid = threadIdx.x;
  const int lane = tid & 31;
  const int wave = tid >> 5;
  const int half = lane >> 4;
  const int r = lane & 15;
  const int bh = blockIdx.y;
  const int b = bh >> 4, h = bh & 15;
  const int q0 = (blockIdx.x * 4 + wave) * 16;

  const uint16_t* Q = qb + (size_t)bh * SEQ * HD;
  const uint16_t* K = kb + (size_t)bh * SEQ * HD;
  const uint16_t* V = vT + (size_t)bh * HD * SEQ;

  // stage: K tile 4KB + V tile 4KB = 512 x 16B chunks, 4 per thread
  auto stage = [&](int buf, int k0) {
#pragma unroll
    for (int c = tid; c < 256; c += 128) {
      const int row = c >> 3, col = (c & 7) * 8;
      async_load_b128(&Kt[buf][row][col], K + (size_t)(k0 + row) * HD + col);
    }
#pragma unroll
    for (int c = tid; c < 256; c += 128) {
      const int d = c >> 2, col = (c & 3) * 8;
      async_load_b128(&Vt[buf][d][col], V + (size_t)d * SEQ + k0 + col);
    }
  };

  stage(0, 0);  // prologue: tile 0 in flight while we set up Q

  const v16h aQ0 = load_fragA(Q + (size_t)(q0 + r) * HD, half);
  const v16h aQ1 = load_fragA(Q + (size_t)(q0 + r) * HD + 32, half);

  v8f acc[4];
#pragma unroll
  for (int t = 0; t < 4; ++t)
#pragma unroll
    for (int e = 0; e < 8; ++e) acc[t][e] = 0.f;
  float mrow[8], lrow[8];
#pragma unroll
  for (int i = 0; i < 8; ++i) { mrow[i] = -3.0e38f; lrow[i] = 0.f; }

  for (int k0 = 0; k0 < SEQ; k0 += 32) {
    const int cur = (k0 >> 5) & 1;
    // Drain our async loads for tile `cur`; the barrier also proves every
    // wave finished reading buffer cur^1 last iteration, so it is safe to
    // refill it with the next tile (DMA overlaps the WMMAs below).
    wait_async0();
    __syncthreads();
    if (k0 + 32 < SEQ) stage(cur ^ 1, k0 + 32);

    // --- S = Q @ K^T for 32 keys (two 16x16 tiles), fragments from LDS ---
    v8f S[2];
#pragma unroll
    for (int sub = 0; sub < 2; ++sub) {
      v16h bK0 = load_fragB(&Kt[cur][sub * 16 + r][0], half);
      v16h bK1 = load_fragB(&Kt[cur][sub * 16 + r][32], half);
      v8f sacc;
#pragma unroll
      for (int e = 0; e < 8; ++e) sacc[e] = 0.f;
      sacc = wmma16(aQ0, bK0, sacc);
      sacc = wmma16(aQ1, bK1, sacc);
      S[sub] = sacc;
    }
    // --- online softmax (row M=i+8*half spans the 16 lanes of a half) ---
    float scl[8];
#pragma unroll
    for (int i = 0; i < 8; ++i) {
      float mx = half_reduce_max(fmaxf(S[0][i], S[1][i]));
      float mnew = fmaxf(mrow[i], mx);
      scl[i] = __expf(mrow[i] - mnew);
      mrow[i] = mnew;
      float p0 = __expf(S[0][i] - mnew);
      float p1 = __expf(S[1][i] - mnew);
      S[0][i] = p0;
      S[1][i] = p1;
      lrow[i] = lrow[i] * scl[i] + half_reduce_sum(p0 + p1);
    }
#pragma unroll
    for (int t = 0; t < 4; ++t)
#pragma unroll
      for (int i = 0; i < 8; ++i) acc[t][i] *= scl[i];
    // --- P: C-layout -> LDS -> A-fragment (same wave; LDS in-order) ---
#pragma unroll
    for (int sub = 0; sub < 2; ++sub)
#pragma unroll
      for (int i = 0; i < 8; ++i)
        Plds[wave][i + 8 * half][sub * 16 + r] = f32_to_f16b(S[sub][i]);
    asm volatile("" ::: "memory");
    FragU aP;
#pragma unroll
    for (int j = 0; j < 8; ++j)
      aP.u[j] =
          *reinterpret_cast<const uint32_t*>(&Plds[wave][r][kpairA(j, half)]);
    // --- O += P @ V (V tile in LDS as [d][key]) ---
#pragma unroll
    for (int t = 0; t < 4; ++t) {
      v16h bV = load_fragB(&Vt[cur][t * 16 + r][0], half);
      acc[t] = wmma16(aP.v, bV, acc[t]);
    }
  }
  // normalize and write O as [b,s,h*64+d] f16
#pragma unroll
  for (int t = 0; t < 4; ++t)
#pragma unroll
    for (int i = 0; i < 8; ++i) {
      const int s = q0 + i + 8 * half;
      ob[((size_t)(b * SEQ + s)) * DM + h * HD + t * 16 + r] =
          f32_to_f16b(acc[t][i] / lrow[i]);
    }
}

// ---- Kernel 3: output projection + bias (f32 out), 32x64 tile per wave ----
__global__ __launch_bounds__(128) void out_gemm(
    const uint16_t* __restrict__ ob, const uint16_t* __restrict__ wo,
    const float* __restrict__ bias, float* __restrict__ out) {
  const int lane = threadIdx.x & 31;
  const int wave = threadIdx.x >> 5;
  const int half = lane >> 4;
  const int r = lane & 15;
  const int tile = blockIdx.x * 4 + wave;
  const int NT = DM / 64;  // 16
  const int mt = tile / NT, nt = tile % NT;
  const int m0 = mt * 32, n0 = nt * 64;

  v8f acc[2][4];
#pragma unroll
  for (int p = 0; p < 2; ++p)
#pragma unroll
    for (int t = 0; t < 4; ++t)
#pragma unroll
      for (int e = 0; e < 8; ++e) acc[p][t][e] = 0.f;

  const uint16_t* arow0 = ob + (size_t)(m0 + r) * DM;
  const uint16_t* arow1 = ob + (size_t)(m0 + 16 + r) * DM;
  for (int k0 = 0; k0 < DM; k0 += 32) {
    __builtin_prefetch(arow0 + k0 + 32, 0, 0);
    __builtin_prefetch(arow1 + k0 + 32, 0, 0);
    v16h a0 = load_fragA(arow0 + k0, half);
    v16h a1 = load_fragA(arow1 + k0, half);
#pragma unroll
    for (int t = 0; t < 4; ++t) {
      v16h b = load_fragB(wo + (size_t)(n0 + t * 16 + r) * DM + k0, half);
      acc[0][t] = wmma16(a0, b, acc[0][t]);
      acc[1][t] = wmma16(a1, b, acc[1][t]);
    }
  }
#pragma unroll
  for (int p = 0; p < 2; ++p)
#pragma unroll
    for (int t = 0; t < 4; ++t) {
      const float bv = bias[n0 + t * 16 + r];
#pragma unroll
      for (int i = 0; i < 8; ++i)
        out[(size_t)(m0 + p * 16 + i + 8 * half) * DM + n0 + t * 16 + r] =
            acc[p][t][i] + bv;
    }
}

extern "C" void kernel_launch(void* const* d_in, const int* in_sizes, int n_in,
                              void* d_out, int out_size, void* d_ws,
                              size_t ws_size, hipStream_t stream) {
  (void)in_sizes; (void)n_in; (void)out_size; (void)ws_size;
  const float* x  = (const float*)d_in[0];
  const float* Wq = (const float*)d_in[1];
  const float* bq = (const float*)d_in[2];
  const float* Wo = (const float*)d_in[3];
  const float* bo = (const float*)d_in[4];
  float* out = (float*)d_out;

  char* ws = (char*)d_ws;
  size_t off = 0;
  auto carve = [&](size_t elems) {
    uint16_t* p = (uint16_t*)(ws + off);
    off += (elems * sizeof(uint16_t) + 255) & ~(size_t)255;
    return p;
  };
  uint16_t* xb   = carve((size_t)ROWS * DM);           // x in f16
  uint16_t* wqb  = carve((size_t)NQKV * DM);           // W_qkv f16
  uint16_t* wob  = carve((size_t)DM * DM);             // W_out f16
  uint16_t* qbuf = carve((size_t)BSZ * NH * SEQ * HD); // Q [b,h,s,d]
  uint16_t* kbuf = carve((size_t)BSZ * NH * SEQ * HD); // K [b,h,s,d]
  uint16_t* vbuf = carve((size_t)BSZ * NH * SEQ * HD); // V^T [b,h,d,s]
  uint16_t* obuf = carve((size_t)ROWS * DM);           // attn out f16

  auto cvt = [&](const float* src, uint16_t* dst, int n) {
    cvt_f32_f16<<<(n + 255) / 256, 256, 0, stream>>>(src, dst, n);
  };
  cvt(x, xb, ROWS * DM);
  cvt(Wq, wqb, NQKV * DM);
  cvt(Wo, wob, DM * DM);

  qkv_gemm_rope<<<(ROWS / 32) * (NQKV / 64) / 4, 128, 0, stream>>>(
      xb, wqb, bq, qbuf, kbuf, vbuf);
  flash_attn<<<dim3(SEQ / 16 / 4, BSZ * NH), 128, 0, stream>>>(qbuf, kbuf,
                                                               vbuf, obuf);
  out_gemm<<<(ROWS / 32) * (DM / 64) / 4, 128, 0, stream>>>(obuf, wob, bo,
                                                            out);
}
